// TransformerLayer_73065983639707
// MI455X (gfx1250) — matrix operand order repaired and checked by reading, not compiled, not run on hardware
//
#include <hip/hip_runtime.h>
#include <math.h>

// ---------------- problem constants ----------------
#define Ee    512
#define Hh    2048
#define Ss    1024
#define Bb    2
#define Pp    128
#define SEXT  (Ss + 2*Pp + 16)   // 1296: padded K/V rows + 16 slack rows
#define WIN   (2*Pp + 16)        // 272: union key window for a 16-query tile
#define WINP  288                // 272 padded up to 9 * 32 for the WMMA K loop

typedef __attribute__((ext_vector_type(16))) __bf16 bf16x16;
typedef __attribute__((ext_vector_type(8)))  float  f32x8;

union FragU { bf16x16 v; uint4 q[2]; };
union H8    { uint4 u; __bf16 h[8]; };

// ---------------- WMMA helpers ----------------
__device__ __forceinline__ f32x8 wmma_bf16(bf16x16 a, bf16x16 b, f32x8 c) {
  return __builtin_amdgcn_wmma_f32_16x16x32_bf16(false, a, false, b, (short)0, c, false, false);
}

// A-matrix 16x32 fragment (ISA layout): lanes 0-15 row m hold K {c..c+7, 16+c..16+c+7}
// with c=0; lanes 16-31 same row, c=8.
__device__ __forceinline__ bf16x16 load_a_frag(const __bf16* base, int ld, int lane) {
  const int m  = lane & 15;
  const int c0 = (lane < 16) ? 0 : 8;
  const __bf16* p = base + (size_t)m * ld;
  FragU f;
  f.q[0] = *(const uint4*)(p + c0);
  f.q[1] = *(const uint4*)(p + c0 + 16);
  return f.v;
}

// B-matrix 32x16 fragment: lane holds column n = lane&15; lanes 0-15 K {0..7, 16..23},
// lanes 16-31 K {8..15, 24..31}.  "base" points at [n0][k0] of an N x K row-major array.
__device__ __forceinline__ bf16x16 load_b_frag(const __bf16* base, int ld, int lane) {
  const int n  = lane & 15;
  const int c0 = (lane < 16) ? 0 : 8;
  const __bf16* p = base + (size_t)n * ld + c0;
  FragU f;
  f.q[0] = *(const uint4*)(p);
  f.q[1] = *(const uint4*)(p + 16);
  return f.v;
}

// async copy of 16 bytes global->LDS (tracked by ASYNCcnt); INST_OFFSET is added
// to both the LDS and the global address per the ISA pseudocode.
__device__ __forceinline__ void async_g2l_16(unsigned lds_addr, const void* gaddr) {
  asm volatile("global_load_async_to_lds_b128 %0, %1, off"
               :: "v"(lds_addr), "v"((unsigned long long)gaddr) : "memory");
}
__device__ __forceinline__ void async_g2l_16_off16(unsigned lds_addr, const void* gaddr) {
  asm volatile("global_load_async_to_lds_b128 %0, %1, off offset:16"
               :: "v"(lds_addr), "v"((unsigned long long)gaddr) : "memory");
}
__device__ __forceinline__ void wait_async0() {
  asm volatile("s_wait_asynccnt 0x0" ::: "memory");
}
__device__ __forceinline__ unsigned lds_addr_of(const void* p) {
  return (unsigned)(uintptr_t)p;   // low 32 bits of a generic LDS pointer = LDS byte offset
}

// ---------------- kernel 1: positional encoding + bf16 copy ----------------
__global__ __launch_bounds__(256) void pe_kernel(const float* __restrict__ x,
                                                 float* __restrict__ xpe,
                                                 __bf16* __restrict__ xbf) {
  const int row = blockIdx.x;          // b*Ss + s
  const int s   = row & (Ss - 1);
  const int tid = threadIdx.x;
  const float kc = -logf(10000.f) / (float)Ee;
  for (int e = tid; e < Ee; e += 256) {
    const int   i2  = (e >> 1) << 1;
    const float ang = (float)s * __expf((float)i2 * kc);
    const float pe  = (e & 1) ? __cosf(ang) : __sinf(ang);
    const float v   = x[(size_t)row * Ee + e] + pe;
    xpe[(size_t)row * Ee + e] = v;
    xbf[(size_t)row * Ee + e] = (__bf16)v;
  }
}

// ---------------- kernel 2: fp32 (K x N) -> bf16 transposed (N x K) ----------------
// K = 1 << kshift (power of two), avoids integer division.
__global__ __launch_bounds__(256) void convert_transpose(const float* __restrict__ w,
                                                         __bf16* __restrict__ wT,
                                                         int kshift, int N, int total) {
  const int idx = blockIdx.x * 256 + threadIdx.x;
  if (idx >= total) return;
  const int n = idx >> kshift;
  const int k = idx & ((1 << kshift) - 1);
  wT[idx] = (__bf16)w[(size_t)k * N + n];
}

// ---------------- kernel 3: fill extended K/V rows with bias ----------------
__global__ __launch_bounds__(256) void fill_bias(__bf16* __restrict__ dst,
                                                 const float* __restrict__ bias,
                                                 int rows) {
  const size_t idx = (size_t)blockIdx.x * 256 + threadIdx.x;
  if (idx >= (size_t)rows * Ee) return;
  dst[idx] = (__bf16)bias[idx & (Ee - 1)];
}

// ---------------- kernel 4: generic bf16 GEMM, 128x128 block tile ----------------
// C(M,N) = A(M,K) * Bt(N,K)^T + bias.  Double-buffered LDS tiles filled with
// ASYNCcnt-tracked global->LDS copies: tile t+1 is in flight while tile t is
// consumed by the WMMA pipe.  Epilogue: optional relu, scale, fp32/bf16 stores,
// pow2 row remap (routes K/V rows into the padded extended buffers).
__global__ __launch_bounds__(256) void gemm_bf16_kernel(
    const __bf16* __restrict__ A, int lda,
    const __bf16* __restrict__ Bt, int ldb,
    const float* __restrict__ bias,
    float* __restrict__ outF, __bf16* __restrict__ outB, int ldo,
    int K,
    int rowShiftIn, int rowsOut, int padOff,
    float scale, int relu) {
  __shared__ __align__(32) __bf16 sA[2][128][32];
  __shared__ __align__(32) __bf16 sB[2][128][32];

  const int tid  = threadIdx.x;
  const int lane = tid & 31;
  const int wave = tid >> 5;
  const int wm   = wave & 3;           // 4 waves along M (32 rows each)
  const int wn   = wave >> 2;          // 2 waves along N (64 cols each)
  const int bm   = blockIdx.y * 128;
  const int bn   = blockIdx.x * 128;

  f32x8 acc[2][4];
#pragma unroll
  for (int i = 0; i < 2; ++i)
#pragma unroll
    for (int j = 0; j < 4; ++j)
      acc[i][j] = (f32x8){0.f,0.f,0.f,0.f,0.f,0.f,0.f,0.f};

  const int lr = tid >> 1;             // 0..127
  const int lc = (tid & 1) * 16;       // 0 or 16
  const unsigned ldsA[2] = { lds_addr_of(&sA[0][lr][lc]), lds_addr_of(&sA[1][lr][lc]) };
  const unsigned ldsB[2] = { lds_addr_of(&sB[0][lr][lc]), lds_addr_of(&sB[1][lr][lc]) };
  const __bf16* ga = A  + (size_t)(bm + lr) * lda + lc;
  const __bf16* gb = Bt + (size_t)(bn + lr) * ldb + lc;

  // prologue: stage tile 0 into buffer 0
  async_g2l_16      (ldsA[0], ga);
  async_g2l_16_off16(ldsA[0], ga);
  async_g2l_16      (ldsB[0], gb);
  async_g2l_16_off16(ldsB[0], gb);
  wait_async0();
  __syncthreads();

  const int T = K >> 5;
  for (int t = 0; t < T; ++t) {
    const int p = t & 1;
    if (t + 1 < T) {
      // prefetch tile t+1 into the other buffer; its last readers finished at
      // the barrier that ended iteration t-1, so the write is race-free.
      const unsigned na = ldsA[p ^ 1], nb = ldsB[p ^ 1];
      const __bf16* gan = ga + (size_t)(t + 1) * 32;
      const __bf16* gbn = gb + (size_t)(t + 1) * 32;
      async_g2l_16      (na, gan);
      async_g2l_16_off16(na, gan);
      async_g2l_16      (nb, gbn);
      async_g2l_16_off16(nb, gbn);
    }

    bf16x16 afr[2], bfr[4];
#pragma unroll
    for (int i = 0; i < 2; ++i) afr[i] = load_a_frag(&sA[p][wm*32 + i*16][0], 32, lane);
#pragma unroll
    for (int j = 0; j < 4; ++j) bfr[j] = load_b_frag(&sB[p][wn*64 + j*16][0], 32, lane);
#pragma unroll
    for (int i = 0; i < 2; ++i)
#pragma unroll
      for (int j = 0; j < 4; ++j)
        acc[i][j] = wmma_bf16(afr[i], bfr[j], acc[i][j]);

    if (t + 1 < T) {
      wait_async0();        // tile t+1 resident
      __syncthreads();      // + all waves done reading buffer p
    }
  }

  const int rowMaskIn = (1 << rowShiftIn) - 1;
#pragma unroll
  for (int i = 0; i < 2; ++i) {
#pragma unroll
    for (int j = 0; j < 4; ++j) {
      const int col = bn + wn*64 + j*16 + (lane & 15);
      const float bv = bias ? bias[col] : 0.f;
#pragma unroll
      for (int r = 0; r < 8; ++r) {
        const int row = bm + wm*32 + i*16 + ((lane >> 4) << 3) + r;
        float v = acc[i][j][r] + bv;
        if (relu) v = fmaxf(v, 0.f);
        v *= scale;
        const size_t orow = (size_t)(row >> rowShiftIn) * rowsOut + padOff + (row & rowMaskIn);
        if (outF) outF[orow * (size_t)ldo + col] = v;
        if (outB) outB[orow * (size_t)ldo + col] = (__bf16)v;
      }
    }
  }
}

// ---------------- kernel 5: windowed attention, 16 queries per block ----------------
__global__ __launch_bounds__(256) void attn_kernel(const __bf16* __restrict__ qs,
                                                   const __bf16* __restrict__ kext,
                                                   const __bf16* __restrict__ vext,
                                                   float* __restrict__ attn_out) {
  __shared__ __align__(32) float  ener[16][WIN];    // 16 x 272 energies
  __shared__ __align__(32) __bf16 wgt[16][WINP];    // 16 x 288 softmax weights (bf16)
  __shared__ __align__(32) __bf16 vT[Ee][32];       // transposed 32-row V slice
  __shared__ float red[16][16];
  __shared__ float rowst[16];

  const int b    = blockIdx.y;
  const int s0   = blockIdx.x * 16;
  const int tid  = threadIdx.x;
  const int lane = tid & 31;
  const int wave = tid >> 5;

  const __bf16* qb = qs   + ((size_t)b * Ss   + s0) * Ee;
  const __bf16* kb = kext + ((size_t)b * SEXT + s0) * Ee;
  const __bf16* vb = vext + ((size_t)b * SEXT + s0) * Ee;

  // ---- energies: (16 x 272) = Qtile(16x512) . Kwin^T, 17 WMMA n-tiles ----
  const int nt[3] = { wave, wave + 8, 16 };
  const int cnt   = (wave == 0) ? 3 : 2;
  f32x8 eacc[3];
  for (int t = 0; t < 3; ++t) eacc[t] = (f32x8){0.f,0.f,0.f,0.f,0.f,0.f,0.f,0.f};

  for (int k0 = 0; k0 < Ee; k0 += 32) {
    if (k0 + 32 < Ee) {
      __builtin_prefetch(qb + k0 + 32 + (size_t)(lane & 15) * Ee, 0, 1);
      __builtin_prefetch(kb + (size_t)(nt[0] * 16 + (lane & 15)) * Ee + k0 + 32, 0, 1);
    }
    const bf16x16 a = load_a_frag(qb + k0, Ee, lane);
    for (int t = 0; t < cnt; ++t) {
      const bf16x16 bb = load_b_frag(kb + (size_t)(nt[t] * 16) * Ee + k0, Ee, lane);
      eacc[t] = wmma_bf16(a, bb, eacc[t]);
    }
  }
  for (int t = 0; t < cnt; ++t) {
    const int col = nt[t]*16 + (lane & 15);
#pragma unroll
    for (int r = 0; r < 8; ++r)
      ener[((lane >> 4) << 3) + r][col] = eacc[t][r];
  }
  __syncthreads();

  // ---- masked softmax: query i sees window j in [i, i+2P] ----
  const int rr = tid >> 4;
  const int cc = tid & 15;
  float mx = -3.0e38f;
  for (int t = 0; t < 17; ++t) {
    const int j = cc + 16*t;
    if (j >= rr && j <= rr + 2*Pp) mx = fmaxf(mx, ener[rr][j]);
  }
  red[rr][cc] = mx;
  __syncthreads();
  if (cc == 0) {
    float m = red[rr][0];
    for (int i = 1; i < 16; ++i) m = fmaxf(m, red[rr][i]);
    rowst[rr] = m;
  }
  __syncthreads();
  const float rmax = rowst[rr];
  float sm = 0.f;
  for (int t = 0; t < 17; ++t) {
    const int j = cc + 16*t;
    if (j >= rr && j <= rr + 2*Pp) sm += __expf(ener[rr][j] - rmax);
  }
  __syncthreads();
  red[rr][cc] = sm;
  __syncthreads();
  if (cc == 0) {
    float s = 0.f;
    for (int i = 0; i < 16; ++i) s += red[rr][i];
    rowst[rr] = s;
  }
  __syncthreads();
  const float rinv = 1.f / rowst[rr];
  for (int t = 0; t < 17; ++t) {
    const int j = cc + 16*t;
    const bool ok = (j >= rr && j <= rr + 2*Pp);
    wgt[rr][j] = ok ? (__bf16)(__expf(ener[rr][j] - rmax) * rinv) : (__bf16)0.f;
  }
  wgt[rr][WIN + cc] = (__bf16)0.f;     // zero-pad 272..287 so padded K-steps are inert
  __syncthreads();

  // ---- output: (16 x 512) = W(16x288) . Vwin(288x512), V staged transposed in LDS ----
  f32x8 oacc[4];
#pragma unroll
  for (int j = 0; j < 4; ++j) oacc[j] = (f32x8){0.f,0.f,0.f,0.f,0.f,0.f,0.f,0.f};

  for (int k0 = 0; k0 < WINP; k0 += 32) {
    // stage vT[n][kk] = Vwin[k0+kk][n]
    for (int idx = tid; idx < 32 * (Ee / 8); idx += 256) {
      const int kk = idx >> 6;            // 0..31
      const int c8 = (idx & 63) << 3;     // column group of 8
      H8 h8;
      h8.u = *(const uint4*)(vb + (size_t)(k0 + kk) * Ee + c8);
#pragma unroll
      for (int j = 0; j < 8; ++j) vT[c8 + j][kk] = h8.h[j];
    }
    __syncthreads();
    const bf16x16 a = load_a_frag(&wgt[0][k0], WINP, lane);
#pragma unroll
    for (int j = 0; j < 4; ++j) {
      const bf16x16 bb = load_b_frag(&vT[(wave*4 + j)*16][0], 32, lane);
      oacc[j] = wmma_bf16(a, bb, oacc[j]);
    }
    __syncthreads();
  }

#pragma unroll
  for (int j = 0; j < 4; ++j) {
    const int col = (wave*4 + j)*16 + (lane & 15);
#pragma unroll
    for (int r = 0; r < 8; ++r) {
      const int row = ((lane >> 4) << 3) + r;
      attn_out[((size_t)b * Ss + s0 + row) * Ee + col] = oacc[j][r];
    }
  }
}

// ---------------- kernel 6: residual add + layernorm ----------------
__global__ __launch_bounds__(256) void ln_kernel(const float* __restrict__ a,
                                                 const float* __restrict__ res,
                                                 const float* __restrict__ g,
                                                 const float* __restrict__ be,
                                                 float* __restrict__ outF,
                                                 __bf16* __restrict__ outB) {
  __shared__ float red[256];
  const int row = blockIdx.x;
  const int tid = threadIdx.x;
  const size_t base = (size_t)row * Ee;
  const float v0 = a[base + tid]       + res[base + tid];
  const float v1 = a[base + tid + 256] + res[base + tid + 256];
  red[tid] = v0 + v1;
  __syncthreads();
  for (int s = 128; s > 0; s >>= 1) { if (tid < s) red[tid] += red[tid + s]; __syncthreads(); }
  const float mu = red[0] * (1.f / Ee);
  __syncthreads();
  const float d0 = v0 - mu, d1 = v1 - mu;
  red[tid] = d0*d0 + d1*d1;
  __syncthreads();
  for (int s = 128; s > 0; s >>= 1) { if (tid < s) red[tid] += red[tid + s]; __syncthreads(); }
  const float rstd = rsqrtf(red[0] * (1.f / Ee) + 1e-5f);
  const float y0 = d0 * rstd * g[tid]       + be[tid];
  const float y1 = d1 * rstd * g[tid + 256] + be[tid + 256];
  if (outF) { outF[base + tid] = y0; outF[base + tid + 256] = y1; }
  if (outB) { outB[base + tid] = (__bf16)y0; outB[base + tid + 256] = (__bf16)y1; }
}

// ---------------- host launcher ----------------
extern "C" void kernel_launch(void* const* d_in, const int* in_sizes, int n_in,
                              void* d_out, int out_size, void* d_ws, size_t ws_size,
                              hipStream_t stream) {
  (void)in_sizes; (void)n_in; (void)out_size; (void)ws_size;

  const float* x    = (const float*)d_in[0];
  const float* wq   = (const float*)d_in[1];
  const float* bq   = (const float*)d_in[2];
  const float* wk   = (const float*)d_in[3];
  const float* bk   = (const float*)d_in[4];
  const float* wv   = (const float*)d_in[5];
  const float* bv   = (const float*)d_in[6];
  const float* w1   = (const float*)d_in[7];
  const float* b1   = (const float*)d_in[8];
  const float* w2   = (const float*)d_in[9];
  const float* b2   = (const float*)d_in[10];
  const float* ln1w = (const float*)d_in[11];
  const float* ln1b = (const float*)d_in[12];
  const float* ln2w = (const float*)d_in[13];
  const float* ln2b = (const float*)d_in[14];
  float* out = (float*)d_out;

  char* ws = (char*)d_ws;
  size_t off = 0;
  auto alloc = [&](size_t bytes) -> void* {
    void* p = ws + off;
    off = (off + bytes + 255) & ~(size_t)255;
    return p;
  };

  const int TOK = Bb * Ss;                               // 2048 rows
  float*  xpe  = (float*) alloc((size_t)TOK * Ee * 4);
  __bf16* xbf  = (__bf16*)alloc((size_t)TOK * Ee * 2);
  __bf16* wqT  = (__bf16*)alloc((size_t)Ee  * Ee * 2);
  __bf16* wkT  = (__bf16*)alloc((size_t)Ee  * Ee * 2);
  __bf16* wvT  = (__bf16*)alloc((size_t)Ee  * Ee * 2);
  __bf16* w1T  = (__bf16*)alloc((size_t)Hh  * Ee * 2);   // (N=H) x (K=E)
  __bf16* w2T  = (__bf16*)alloc((size_t)Ee  * Hh * 2);   // (N=E) x (K=H)
  __bf16* qsc  = (__bf16*)alloc((size_t)TOK * Ee * 2);   // scaled Q, bf16
  __bf16* kext = (__bf16*)alloc((size_t)Bb * SEXT * Ee * 2);
  __bf16* vext = (__bf16*)alloc((size_t)Bb * SEXT * Ee * 2);
  float*  attn = (float*) alloc((size_t)TOK * Ee * 4);
  float*  x1   = (float*) alloc((size_t)TOK * Ee * 4);
  __bf16* x1bf = (__bf16*)alloc((size_t)TOK * Ee * 2);
  __bf16* hbuf = (__bf16*)alloc((size_t)TOK * Hh * 2);
  float*  ffb  = (float*) alloc((size_t)TOK * Ee * 4);

  auto cdiv = [](long a, long b) { return (int)((a + b - 1) / b); };

  // x + positional encoding (fp32 + bf16 copies)
  pe_kernel<<<TOK, 256, 0, stream>>>(x, xpe, xbf);

  // weight conversion / transposition to bf16 (N x K row-major)
  convert_transpose<<<cdiv((long)Ee*Ee, 256), 256, 0, stream>>>(wq, wqT, 9, Ee, Ee*Ee);
  convert_transpose<<<cdiv((long)Ee*Ee, 256), 256, 0, stream>>>(wk, wkT, 9, Ee, Ee*Ee);
  convert_transpose<<<cdiv((long)Ee*Ee, 256), 256, 0, stream>>>(wv, wvT, 9, Ee, Ee*Ee);
  convert_transpose<<<cdiv((long)Hh*Ee, 256), 256, 0, stream>>>(w1, w1T, 9, Hh, Hh*Ee);
  convert_transpose<<<cdiv((long)Ee*Hh, 256), 256, 0, stream>>>(w2, w2T, 11, Ee, Ee*Hh);

  // padded K/V rows start as pure bias (matches x_ext zero-padding @ W + b)
  fill_bias<<<cdiv((long)Bb*SEXT*Ee, 256), 256, 0, stream>>>(kext, bk, Bb*SEXT);
  fill_bias<<<cdiv((long)Bb*SEXT*Ee, 256), 256, 0, stream>>>(vext, bv, Bb*SEXT);

  const float qscale = 0.044194173824159216f;            // 1/sqrt(512)

  // Q (scaled, bf16) ; K/V routed into padded extended buffers
  gemm_bf16_kernel<<<dim3(Ee/128, TOK/128), 256, 0, stream>>>(
      xbf, Ee, wqT, Ee, bq, nullptr, qsc, Ee,
      Ee, 11, TOK, 0, qscale, 0);
  gemm_bf16_kernel<<<dim3(Ee/128, TOK/128), 256, 0, stream>>>(
      xbf, Ee, wkT, Ee, bk, nullptr, kext, Ee,
      Ee, 10, SEXT, Pp, 1.f, 0);
  gemm_bf16_kernel<<<dim3(Ee/128, TOK/128), 256, 0, stream>>>(
      xbf, Ee, wvT, Ee, bv, nullptr, vext, Ee,
      Ee, 10, SEXT, Pp, 1.f, 0);

  // windowed attention
  attn_kernel<<<dim3(Ss/16, Bb), 256, 0, stream>>>(qsc, kext, vext, attn);

  // LN1(attn + x)
  ln_kernel<<<TOK, 256, 0, stream>>>(attn, xpe, ln1w, ln1b, x1, x1bf);

  // FFN
  gemm_bf16_kernel<<<dim3(Hh/128, TOK/128), 256, 0, stream>>>(
      x1bf, Ee, w1T, Ee, b1, nullptr, hbuf, Hh,
      Ee, 11, TOK, 0, 1.f, 1);
  gemm_bf16_kernel<<<dim3(Ee/128, TOK/128), 256, 0, stream>>>(
      hbuf, Hh, w2T, Hh, b2, ffb, nullptr, Ee,
      Hh, 11, TOK, 0, 1.f, 0);

  // LN2(ff + x1) -> output
  ln_kernel<<<TOK, 256, 0, stream>>>(ffb, x1, ln2w, ln2b, out, nullptr);
}
